// Encoder_30253749633264
// MI455X (gfx1250) — matrix operand order repaired and checked by reading, not compiled, not run on hardware
//
#include <hip/hip_runtime.h>
#include <stdint.h>

// ---------------------------------------------------------------------------
// Transformer encoder (SASRec-style) for MI455X / gfx1250, wave32 + WMMA bf16.
// B=8, S=1022 -> L=1024, D=512, H=8, DK=64, F=2048, NL=4.
// Compute path: V_WMMA_F32_16X16X32_BF16 (bf16 A/B, f32 accum).
// GEMM: 256x64 workgroup tile, 8 waves x (32x64), double-buffered LDS.
// A tiles move global->LDS via CDNA5 async DMA (GLOBAL_LOAD_ASYNC_TO_LDS_B128,
// ASYNCcnt), overlapping the WMMA block with zero VGPR staging cost.
// ---------------------------------------------------------------------------

#define B_   8
#define S_   1022
#define L_   1024
#define D_   512
#define H_   8
#define DK_  64
#define F_   2048
#define NL_  4
#define BL_  (B_ * L_)
#define NEGV (-1000000000.0f)

typedef __attribute__((ext_vector_type(16))) __bf16 bf16x16;
typedef __attribute__((ext_vector_type(8)))  float  f32x8;

union alignas(32) Frag {
    bf16x16 v;
    uint4   u[2];
};

// ---------------------------------------------------------------------------
// Generic batched WMMA GEMM:  C = alpha * (A @ B) + bias  [+ mask / relu]
//   A: [M,K] bf16 row-major (lda), batch base = bo*aOut + bi*aIn, batch=z
//   B: TRANS_B ? [N,K] : [K,N] bf16 row-major (ldb)
//   C: f32 or bf16, row-major (ldc)
// Workgroup: 256 threads = 8 waves; tile 256(M) x 64(N); K step 32.
// Wave w: rows [m0+32w, m0+32w+32) x cols [n0, n0+64) = 2x4 wmma tiles.
// ---------------------------------------------------------------------------
template <bool TRANS_B, bool RELU, bool OUT_BF16, bool MASK>
__global__ __launch_bounds__(256, 1) void wmma_gemm(
    const __bf16* __restrict__ A, int lda, long long aOut, long long aIn,
    const __bf16* __restrict__ Bm, int ldb, long long bOut, long long bIn,
    const float* __restrict__ bias,
    float* __restrict__ Cf, __bf16* __restrict__ Cb,
    int ldc, long long cOut, long long cIn,
    int M, int N, int K, float alpha,
    const int* __restrict__ enc, int Hdiv)
{
    __shared__ __bf16 Alds[2][256 * 32];   // A tiles, row-major [256][32]
    __shared__ __bf16 Blds[2][64 * 32];    // B tiles stored transposed: [n][k]

    const int tid  = threadIdx.x;
    const int wid  = tid >> 5;
    const int lane = tid & 31;
    const int half = lane >> 4;   // 0: lanes 0-15, 1: lanes 16-31
    const int lr   = lane & 15;

    const int batch = blockIdx.z;
    const int bo = batch / Hdiv;
    const int bi = batch % Hdiv;
    const size_t aBase = (size_t)bo * aOut + (size_t)bi * aIn;
    const size_t bBase = (size_t)bo * bOut + (size_t)bi * bIn;
    const size_t cBase = (size_t)bo * cOut + (size_t)bi * cIn;

    const int m0 = blockIdx.y * 256;
    const int n0 = blockIdx.x * 64;

    f32x8 acc[2][4] = {};

    uint4 bReg;      // staged B: 64x32 bf16 = 256 uint4 / 256 thr = 1 each

    // A tile: 256x32 bf16 = 16 KB = 4 x 16B chunks per thread, moved with the
    // CDNA5 async DMA path (global -> LDS, tracked by ASYNCcnt, no VGPR data).
    auto asyncLoadA = [&](int kk, int bufSel) {
#pragma unroll
        for (int i = 0; i < 4; ++i) {
            int c   = tid + i * 256;      // 0..1023
            int row = c >> 2;
            int off = (c & 3) << 3;
            unsigned long long gaddr = (unsigned long long)(const void*)
                &A[aBase + (size_t)(m0 + row) * lda + kk + off];
            // low 32 bits of the LDS-aperture flat address == LDS byte address
            unsigned int laddr = (unsigned int)(unsigned long long)(void*)
                &Alds[bufSel][row * 32 + off];
            asm volatile("global_load_async_to_lds_b128 %0, %1, off"
                         :: "v"(laddr), "v"(gaddr) : "memory");
        }
    };

    auto loadB = [&](int kk) {
        if constexpr (TRANS_B) {
            int n   = tid >> 2;
            int off = (tid & 3) << 3;
            bReg = *(const uint4*)&Bm[bBase + (size_t)(n0 + n) * ldb + kk + off];
        } else {
            int krow = tid >> 3;          // 0..31
            int nc0  = (tid & 7) << 3;    // 0..56
            bReg = *(const uint4*)&Bm[bBase + (size_t)(kk + krow) * ldb + n0 + nc0];
        }
    };

    auto storeB = [&](int bufSel) {
        if constexpr (TRANS_B) {
            int n   = tid >> 2;
            int off = (tid & 3) << 3;
            *(uint4*)&Blds[bufSel][n * 32 + off] = bReg;
        } else {
            int krow = tid >> 3;
            int nc0  = (tid & 7) << 3;
            const __bf16* tp = (const __bf16*)&bReg;
#pragma unroll
            for (int j = 0; j < 8; ++j)
                Blds[bufSel][(nc0 + j) * 32 + krow] = tp[j];
        }
    };

    asyncLoadA(0, 0);
    loadB(0);
    storeB(0);
    asm volatile("s_wait_asynccnt 0x0" ::: "memory");
    __syncthreads();

    int buf = 0;
    for (int kk = 0; kk < K; kk += 32) {
        const bool hasNext = (kk + 32 < K);
        if (hasNext) {
            asyncLoadA(kk + 32, buf ^ 1);            // DMA overlaps WMMA below
            loadB(kk + 32);
            if (kk + 64 < K)
                __builtin_prefetch(&A[aBase + (size_t)(m0 + tid) * lda + kk + 64], 0, 1);
        }

        // A fragments: lanes0-15 hold K 0-7,16-23; lanes16-31 hold K 8-15,24-31
        Frag a[2];
#pragma unroll
        for (int mt = 0; mt < 2; ++mt) {
            const __bf16* ap = &Alds[buf][(wid * 32 + mt * 16 + lr) * 32];
            a[mt].u[0] = *(const uint4*)(ap + half * 8);
            a[mt].u[1] = *(const uint4*)(ap + 16 + half * 8);
        }
#pragma unroll
        for (int nt = 0; nt < 4; ++nt) {
            // B fragment: lane = column, lanes0-15 hold K 0-7,16-23; lanes16-31 K 8-15,24-31
            Frag b;
            const __bf16* bp = &Blds[buf][(nt * 16 + lr) * 32 + half * 8];
            b.u[0] = *(const uint4*)(bp);
            b.u[1] = *(const uint4*)(bp + 16);
#pragma unroll
            for (int mt = 0; mt < 2; ++mt)
                acc[mt][nt] = __builtin_amdgcn_wmma_f32_16x16x32_bf16(
                    false, a[mt].v, false, b.v, (short)0, acc[mt][nt], false, false);
        }

        if (hasNext) {
            storeB(buf ^ 1);
            asm volatile("s_wait_asynccnt 0x0" ::: "memory");
            __syncthreads();
            buf ^= 1;
        }
    }

    // ---- epilogue: D layout VGPR e holds row (half*8 + e), col = lane&15
#pragma unroll
    for (int nt = 0; nt < 4; ++nt) {
        int col = n0 + nt * 16 + lr;
        float bv = bias ? bias[col] : 0.0f;
        bool masked = false;
        if constexpr (MASK)
            masked = (col >= 2) && (enc[(size_t)bo * S_ + (col - 2)] == 0);
#pragma unroll
        for (int mt = 0; mt < 2; ++mt) {
#pragma unroll
            for (int e = 0; e < 8; ++e) {
                int row = m0 + wid * 32 + mt * 16 + half * 8 + e;
                float v = acc[mt][nt][e] * alpha + bv;
                if constexpr (MASK) { if (masked) v = NEGV; }
                if constexpr (RELU) v = fmaxf(v, 0.0f);
                size_t idx = cBase + (size_t)row * ldc + col;
                if constexpr (OUT_BF16) Cb[idx] = (__bf16)v;
                else                    Cf[idx] = v;
            }
        }
    }
}

// ---------------------------------------------------------------------------
// Row softmax over masked scores (in-place bf16) + mean over heads -> f32 out.
// One block per (b, q); loops all H heads so the mean needs no atomics.
// ---------------------------------------------------------------------------
__global__ __launch_bounds__(256) void softmax_mean_kernel(
    __bf16* __restrict__ probs, float* __restrict__ meanOut)
{
    __shared__ float red[256];
    const int tid = threadIdx.x;
    const int b  = blockIdx.x / L_;
    const int qi = blockIdx.x % L_;
    float mean[4] = {0.f, 0.f, 0.f, 0.f};

    for (int h = 0; h < H_; ++h) {
        __bf16* row = probs + (((size_t)(b * H_ + h)) * L_ + qi) * L_;
        float v[4];
#pragma unroll
        for (int i = 0; i < 4; ++i) v[i] = (float)row[tid + i * 256];

        float m = fmaxf(fmaxf(v[0], v[1]), fmaxf(v[2], v[3]));
        red[tid] = m; __syncthreads();
        for (int s = 128; s > 0; s >>= 1) {
            if (tid < s) red[tid] = fmaxf(red[tid], red[tid + s]);
            __syncthreads();
        }
        m = red[0]; __syncthreads();

        float e[4], sum = 0.f;
#pragma unroll
        for (int i = 0; i < 4; ++i) { e[i] = __expf(v[i] - m); sum += e[i]; }
        red[tid] = sum; __syncthreads();
        for (int s = 128; s > 0; s >>= 1) {
            if (tid < s) red[tid] += red[tid + s];
            __syncthreads();
        }
        float inv = 1.0f / red[0]; __syncthreads();

#pragma unroll
        for (int i = 0; i < 4; ++i) {
            float p = e[i] * inv;
            row[tid + i * 256] = (__bf16)p;
            mean[i] += p * (1.0f / H_);
        }
    }
    float* mrow = meanOut + ((size_t)b * L_ + qi) * L_;
#pragma unroll
    for (int i = 0; i < 4; ++i) mrow[tid + i * 256] = mean[i];
}

// ---------------------------------------------------------------------------
// x = LayerNorm(t + resid) * g + beta ; writes f32 master + bf16 shadow.
// One block per token row (D=512, 2 elements/thread).
// ---------------------------------------------------------------------------
__global__ __launch_bounds__(256) void residual_ln_kernel(
    const float* __restrict__ t, const float* __restrict__ resid,
    const float* __restrict__ g, const float* __restrict__ be,
    float* __restrict__ outF, __bf16* __restrict__ outB)
{
    __shared__ float red[256];
    const int tid = threadIdx.x;
    const size_t base = (size_t)blockIdx.x * D_;

    float v0 = t[base + tid]       + resid[base + tid];
    float v1 = t[base + tid + 256] + resid[base + tid + 256];

    red[tid] = v0 + v1; __syncthreads();
    for (int s = 128; s > 0; s >>= 1) { if (tid < s) red[tid] += red[tid + s]; __syncthreads(); }
    float mu = red[0] * (1.0f / D_); __syncthreads();

    float d0 = v0 - mu, d1 = v1 - mu;
    red[tid] = d0 * d0 + d1 * d1; __syncthreads();
    for (int s = 128; s > 0; s >>= 1) { if (tid < s) red[tid] += red[tid + s]; __syncthreads(); }
    float rs = rsqrtf(red[0] * (1.0f / D_) + 1e-5f);

    float y0 = d0 * rs * g[tid]       + be[tid];
    float y1 = d1 * rs * g[tid + 256] + be[tid + 256];
    outF[base + tid]       = y0;  outF[base + tid + 256] = y1;
    outB[base + tid]       = (__bf16)y0;
    outB[base + tid + 256] = (__bf16)y1;
}

// ---------------------------------------------------------------------------
// Embedding gather (user/item/token) + sinusoidal positional encoding.
// ---------------------------------------------------------------------------
__global__ __launch_bounds__(256) void embed_kernel(
    const float* __restrict__ ue, const float* __restrict__ ie,
    const int* __restrict__ enc, const float* __restrict__ tbl,
    float* __restrict__ xf, __bf16* __restrict__ xb)
{
    const size_t idx = (size_t)blockIdx.x * 256 + threadIdx.x;
    const int d = (int)(idx % D_);
    const int t = (int)((idx / D_) % L_);
    const int b = (int)(idx / ((size_t)D_ * L_));

    float base;
    if      (t == 0) base = ue[(size_t)b * D_ + d];
    else if (t == 1) base = ie[(size_t)b * D_ + d];
    else             base = tbl[(size_t)enc[(size_t)b * S_ + (t - 2)] * D_ + d];

    int   pair = d >> 1;
    float freq = __expf(-9.210340371976184f * (float)(2 * pair) / (float)D_);
    float ang  = (float)t * freq;
    float pe   = (d & 1) ? __cosf(ang) : __sinf(ang);
    float v    = base + pe;
    xf[idx] = v;
    xb[idx] = (__bf16)v;
}

__global__ __launch_bounds__(256) void cvt_bf16_kernel(
    const float* __restrict__ s, __bf16* __restrict__ d)
{
    size_t i = (size_t)blockIdx.x * 256 + threadIdx.x;
    d[i] = (__bf16)s[i];
}

__global__ __launch_bounds__(256) void copy_f32_kernel(
    const float* __restrict__ s, float* __restrict__ d)
{
    size_t i = (size_t)blockIdx.x * 256 + threadIdx.x;
    d[i] = s[i];
}

// ---------------------------------------------------------------------------
extern "C" void kernel_launch(void* const* d_in, const int* in_sizes, int n_in,
                              void* d_out, int out_size, void* d_ws, size_t ws_size,
                              hipStream_t stream)
{
    const float* ue   = (const float*)d_in[0];
    const float* ie   = (const float*)d_in[1];
    const int*   enc  = (const int*)  d_in[2];
    const float* tbl  = (const float*)d_in[3];
    const float* Wq   = (const float*)d_in[4];
    const float* bq   = (const float*)d_in[5];
    const float* Wk   = (const float*)d_in[6];
    const float* bk   = (const float*)d_in[7];
    const float* Wv   = (const float*)d_in[8];
    const float* bv   = (const float*)d_in[9];
    const float* Wo   = (const float*)d_in[10];
    const float* bo   = (const float*)d_in[11];
    const float* ln1g = (const float*)d_in[12];
    const float* ln1b = (const float*)d_in[13];
    const float* W1   = (const float*)d_in[14];
    const float* b1   = (const float*)d_in[15];
    const float* W2   = (const float*)d_in[16];
    const float* b2   = (const float*)d_in[17];
    const float* ln2g = (const float*)d_in[18];
    const float* ln2b = (const float*)d_in[19];
    float* out = (float*)d_out;

    // ---- workspace carve (total = 256 MB exactly) ----
    char* p = (char*)d_ws;
    auto carve = [&](size_t bytes) -> char* {
        char* r = p; p += (bytes + 255) & ~(size_t)255; return r;
    };
    float*  xf    = (float*) carve((size_t)BL_ * D_ * 4);
    __bf16* xb    = (__bf16*)carve((size_t)BL_ * D_ * 2);
    __bf16* qb    = (__bf16*)carve((size_t)BL_ * D_ * 2);
    __bf16* kb    = (__bf16*)carve((size_t)BL_ * D_ * 2);
    __bf16* vb    = (__bf16*)carve((size_t)BL_ * D_ * 2);
    __bf16* ctxb  = (__bf16*)carve((size_t)BL_ * D_ * 2);
    __bf16* hb    = (__bf16*)carve((size_t)BL_ * F_ * 2);
    float*  tmpf  = (float*) carve((size_t)BL_ * D_ * 4);
    __bf16* probs = (__bf16*)carve((size_t)B_ * H_ * L_ * L_ * 2);
    __bf16* wqb   = (__bf16*)carve((size_t)NL_ * D_ * D_ * 2);
    __bf16* wkb   = (__bf16*)carve((size_t)NL_ * D_ * D_ * 2);
    __bf16* wvb   = (__bf16*)carve((size_t)NL_ * D_ * D_ * 2);
    __bf16* wob   = (__bf16*)carve((size_t)NL_ * D_ * D_ * 2);
    __bf16* w1b   = (__bf16*)carve((size_t)NL_ * D_ * F_ * 2);
    __bf16* w2b   = (__bf16*)carve((size_t)NL_ * F_ * D_ * 2);

    // ---- one-time f32 -> bf16 weight conversion ----
    cvt_bf16_kernel<<<NL_ * D_ * D_ / 256, 256, 0, stream>>>(Wq, wqb);
    cvt_bf16_kernel<<<NL_ * D_ * D_ / 256, 256, 0, stream>>>(Wk, wkb);
    cvt_bf16_kernel<<<NL_ * D_ * D_ / 256, 256, 0, stream>>>(Wv, wvb);
    cvt_bf16_kernel<<<NL_ * D_ * D_ / 256, 256, 0, stream>>>(Wo, wob);
    cvt_bf16_kernel<<<NL_ * D_ * F_ / 256, 256, 0, stream>>>(W1, w1b);
    cvt_bf16_kernel<<<NL_ * F_ * D_ / 256, 256, 0, stream>>>(W2, w2b);

    // ---- embeddings + positional encoding ----
    embed_kernel<<<BL_ * D_ / 256, 256, 0, stream>>>(ue, ie, enc, tbl, xf, xb);

    const dim3 blk(256);
    const dim3 gProj(D_ / 64, BL_ / 256, 1);            // 8192x512 GEMMs
    const dim3 gScor(L_ / 64, L_ / 256, B_ * H_);       // batched 1024x1024xK64
    const dim3 gCtx (1,       L_ / 256, B_ * H_);       // batched 1024x64xK1024
    const dim3 gFfn1(F_ / 64, BL_ / 256, 1);            // 8192x2048xK512

    for (int l = 0; l < NL_; ++l) {
        const __bf16* wq_l = wqb + (size_t)l * D_ * D_;
        const __bf16* wk_l = wkb + (size_t)l * D_ * D_;
        const __bf16* wv_l = wvb + (size_t)l * D_ * D_;
        const __bf16* wo_l = wob + (size_t)l * D_ * D_;
        const __bf16* w1_l = w1b + (size_t)l * D_ * F_;
        const __bf16* w2_l = w2b + (size_t)l * F_ * D_;

        // Q/K/V projections: x[8192,512] @ W[512,512] + b -> bf16
        wmma_gemm<false, false, true, false><<<gProj, blk, 0, stream>>>(
            xb, D_, 0, 0, wq_l, D_, 0, 0, bq + l * D_,
            nullptr, qb, D_, 0, 0, BL_, D_, D_, 1.0f, nullptr, 1);
        wmma_gemm<false, false, true, false><<<gProj, blk, 0, stream>>>(
            xb, D_, 0, 0, wk_l, D_, 0, 0, bk + l * D_,
            nullptr, kb, D_, 0, 0, BL_, D_, D_, 1.0f, nullptr, 1);
        wmma_gemm<false, false, true, false><<<gProj, blk, 0, stream>>>(
            xb, D_, 0, 0, wv_l, D_, 0, 0, bv + l * D_,
            nullptr, vb, D_, 0, 0, BL_, D_, D_, 1.0f, nullptr, 1);

        // scores = scale * Q @ K^T, masked on padded key columns -> bf16 probs buf
        wmma_gemm<true, false, true, true><<<gScor, blk, 0, stream>>>(
            qb, D_, (long long)L_ * D_, DK_,
            kb, D_, (long long)L_ * D_, DK_,
            nullptr, nullptr, probs, L_,
            (long long)H_ * L_ * L_, (long long)L_ * L_,
            L_, L_, DK_, 0.125f, enc, H_);

        // softmax in place + mean over heads straight into output slab
        softmax_mean_kernel<<<B_ * L_, blk, 0, stream>>>(
            probs, out + (size_t)BL_ * D_ + (size_t)l * B_ * L_ * L_);

        // ctx = probs @ V  (batched over b,h) -> bf16 [B,L,H*DK]
        wmma_gemm<false, false, true, false><<<gCtx, blk, 0, stream>>>(
            probs, L_, (long long)H_ * L_ * L_, (long long)L_ * L_,
            vb, D_, (long long)L_ * D_, DK_,
            nullptr, nullptr, ctxb, D_, (long long)L_ * D_, DK_,
            L_, DK_, L_, 1.0f, nullptr, H_);

        // attention out projection -> f32 tmp, then residual + LN1
        wmma_gemm<false, false, false, false><<<gProj, blk, 0, stream>>>(
            ctxb, D_, 0, 0, wo_l, D_, 0, 0, bo + l * D_,
            tmpf, nullptr, D_, 0, 0, BL_, D_, D_, 1.0f, nullptr, 1);
        residual_ln_kernel<<<BL_, blk, 0, stream>>>(
            tmpf, xf, ln1g + l * D_, ln1b + l * D_, xf, xb);

        // FFN: relu(x @ W1 + b1) -> bf16 hidden ; hidden @ W2 + b2 -> f32 tmp
        wmma_gemm<false, true, true, false><<<gFfn1, blk, 0, stream>>>(
            xb, D_, 0, 0, w1_l, F_, 0, 0, b1 + l * F_,
            nullptr, hb, F_, 0, 0, BL_, F_, D_, 1.0f, nullptr, 1);
        wmma_gemm<false, false, false, false><<<gProj, blk, 0, stream>>>(
            hb, F_, 0, 0, w2_l, D_, 0, 0, b2 + l * D_,
            tmpf, nullptr, D_, 0, 0, BL_, D_, F_, 1.0f, nullptr, 1);
        residual_ln_kernel<<<BL_, blk, 0, stream>>>(
            tmpf, xf, ln2g + l * D_, ln2b + l * D_, xf, xb);
    }

    // final hidden states -> first section of d_out
    copy_f32_kernel<<<BL_ * D_ / 256, 256, 0, stream>>>(xf, out);
}
